// NestedTriangleMultiplicativeUpdate_56023553409069
// MI455X (gfx1250) — compile-verified, matrix operand order checked by reading, not run on hardware
//
#include <hip/hip_runtime.h>
#include <hip/hip_bf16.h>

typedef __attribute__((ext_vector_type(16))) _Float16 v16h;
typedef __attribute__((ext_vector_type(8)))  float    v8f;
typedef int v4i_vs __attribute__((vector_size(16)));
typedef v4i_vs __attribute__((address_space(1)))* as1_v4i_p;
typedef v4i_vs __attribute__((address_space(3)))* as3_v4i_p;
typedef __attribute__((address_space(1))) void* as1_void_p;
typedef __attribute__((address_space(3))) void* as3_void_p;

#define NN   1024
#define KKE  32
#define CS   384
#define CZ   128
#define CG   16
#define NRBF 64

#if defined(__has_builtin)
#if __has_builtin(__builtin_amdgcn_global_load_async_to_lds_b128)
#define HAVE_ASYNC_LDS 1
#endif
#endif

// ---------------- LDS layout (bytes) ----------------
// sw0 : 32768  (phase A: W_ep swizzled f16 ; phase B: W_og)
// sw1 : 32768  (phase A: W_eg            ; phase B: W_lo)
// swdp: 16384  (W_dp swizzled f16, 16 tiles)
// efL : 16896  (32 x 132 f32; ef after LN, later reused as upd accumulator)
// e2T : 20480  (early: raw edge-feature block [32x128] f32 via async DMA;
//               later: 128 x 40 f32 transposed e2 with padded stride)
// dist:  4224  (32 x 33 f32)
// ogS :  8192  (32 x 128 f16 out-gate)
// tpos/g3/se2/nlr/stat : small
#define OFF_SW0   0
#define OFF_SW1   32768
#define OFF_WDP   65536
#define OFF_EF    81920
#define OFF_E2T   98816
#define OFF_DIST  119296
#define OFF_OG    123520
#define OFF_TPOS  131712
#define OFF_G3    132224
#define OFF_SE2   132736
#define OFF_NLR   133248
#define OFF_STAT  133376
#define SHMEM_BYTES 133632

// Per-lane K index mapping of the 16-bit 16x32 WMMA A/B operand
// (lanes 0-15: K = 0-7 in elems 0-7, 16-23 in elems 8-15;
//  lanes 16-31: K = 8-15 / 24-31).
__device__ __forceinline__ int kmap(int h, int e) {
    int base = (e < 8) ? e : (e + 8);
    return base + 8 * h;
}

__device__ __forceinline__ float sigmoidf_(float x) {
    return 1.0f / (1.0f + __expf(-x));
}

__device__ __forceinline__ v8f wmma16(v16h a, v16h b, v8f c) {
    // v_wmma_f32_16x16x32_f16  D = A(16x32) x B(32x16) + C
    return __builtin_amdgcn_wmma_f32_16x16x32_f16(
        false, a, false, b, (short)0, c, false, false);
}

__device__ __forceinline__ void wait_asynccnt0() {
#if defined(__has_builtin) && __has_builtin(__builtin_amdgcn_s_wait_asynccnt)
    __builtin_amdgcn_s_wait_asynccnt(0);
#else
    asm volatile("s_wait_asynccnt 0x0" ::: "memory");
#endif
}

// Stage a [32*ksTiles x 128] row-major f32 weight matrix into LDS f16,
// pre-swizzled into the per-lane WMMA-B layout: tile t = ks*8+zt, each lane
// owns 16 contiguous halves -> one 32B (2 x ds_load_b128) fetch at use time.
__device__ void stage_w(_Float16* dst, const float* W, int ksTiles, int tid) {
    const int total = ksTiles * 8 * 512;
    for (int idx = tid; idx < total; idx += 256) {
        int t    = idx >> 9;
        int lane = (idx >> 4) & 31;
        int e    = idx & 15;
        int ks   = t >> 3;
        int zt   = t & 7;
        int k    = kmap(lane >> 4, e) + 32 * ks;
        int col  = zt * 16 + (lane & 15);
        dst[idx] = (_Float16)W[k * CZ + col];
    }
}

// Gather one 16x32 f16 A-tile from an f32 LDS matrix [rows][stride].
__device__ __forceinline__ v16h gather_a(const float* src, int stride,
                                         int m, int ks, int lane) {
    v16h r;
    int j = (lane & 15) + 16 * m;
    int h = lane >> 4;
#pragma unroll
    for (int e = 0; e < 16; ++e) {
        int k = kmap(h, e) + 32 * ks;
        r[e] = (_Float16)src[j * stride + k];
    }
    return r;
}

__global__ __launch_bounds__(256) void ntmu_kernel(
    const float* __restrict__ nodef, const float* __restrict__ ntrans,
    const float* __restrict__ edgef, const int* __restrict__ eidx,
    const float* __restrict__ W_nl, const float* __restrict__ b_nl,
    const float* __restrict__ W_nr, const float* __restrict__ b_nr,
    const float* __restrict__ ln_g, const float* __restrict__ ln_b,
    const float* __restrict__ W_ep, const float* __restrict__ b_ep,
    const float* __restrict__ W_eg, const float* __restrict__ b_eg,
    const float* __restrict__ W_dg, const float* __restrict__ b_dg,
    const float* __restrict__ W_dp, const float* __restrict__ b_dp,
    const float* __restrict__ lno_g, const float* __restrict__ lno_b,
    const float* __restrict__ W_lo, const float* __restrict__ b_lo,
    const float* __restrict__ W_og, const float* __restrict__ b_og,
    float* __restrict__ out)
{
    extern __shared__ char smem[];
    _Float16* sw0  = (_Float16*)(smem + OFF_SW0);
    _Float16* sw1  = (_Float16*)(smem + OFF_SW1);
    _Float16* swdp = (_Float16*)(smem + OFF_WDP);
    float*    efL  = (float*)(smem + OFF_EF);     // 32 x 132 (ef, later upd)
    float*    e2T  = (float*)(smem + OFF_E2T);    // 128 x 40 (later)
    float*    edL  = (float*)(smem + OFF_E2T);    // 32 x 128 raw edge rows (early)
    float*    distL= (float*)(smem + OFF_DIST);   // 32 x 33
    _Float16* ogS  = (_Float16*)(smem + OFF_OG);  // 32 x 128
    float*    tpos = (float*)(smem + OFF_TPOS);   // 32 x 4
    float*    g3   = (float*)(smem + OFF_G3);     // 128
    float*    se2  = (float*)(smem + OFF_SE2);    // 128
    float*    nlr  = (float*)(smem + OFF_NLR);    // 32
    float*    stat = (float*)(smem + OFF_STAT);   // 32 x 2

    const int n    = blockIdx.x;
    const int tid  = threadIdx.x;
    const int lane = tid & 31;
    const int w    = tid >> 5;     // wave id 0..7
    const int h    = lane >> 4;    // row-half selector
    const int col  = lane & 15;    // C-matrix column within z-tile

    // -------- kick off async DMA of this node's edge-feature block to LDS ----
#if defined(HAVE_ASYNC_LDS)
#pragma unroll
    for (int it = 0; it < 4; ++it) {
        int idx = tid + it * 256;                 // 1024 x 16B chunks
        __builtin_amdgcn_global_load_async_to_lds_b128(
            (as1_v4i_p)(as1_void_p)(edgef + (size_t)n * KKE * CZ + idx * 4),
            (as3_v4i_p)(as3_void_p)(edL + idx * 4),
            0, 0);
    }
#endif
    // Warm L2 for the phase-B weights used much later.
    __builtin_prefetch(W_og + tid * 64, 0, 1);
    __builtin_prefetch(W_lo + tid * 64, 0, 1);

    // -------- positions of the K neighbors --------
    if (tid < KKE) {
        int d = eidx[NN * KKE + n * KKE + tid];
        tpos[tid * 4 + 0] = ntrans[d * 3 + 0];
        tpos[tid * 4 + 1] = ntrans[d * 3 + 1];
        tpos[tid * 4 + 2] = ntrans[d * 3 + 2];
    }
    __syncthreads();

    // -------- pairwise distances + weight staging + nl/nr --------
    for (int p = tid; p < KKE * KKE; p += 256) {
        int i = p >> 5, j = p & 31;
        float dx = tpos[j * 4 + 0] - tpos[i * 4 + 0] + 1e-8f;
        float dy = tpos[j * 4 + 1] - tpos[i * 4 + 1] + 1e-8f;
        float dz = tpos[j * 4 + 2] - tpos[i * 4 + 2] + 1e-8f;
        distL[i * 33 + j] = sqrtf(dx * dx + dy * dy + dz * dz);
    }
    stage_w(swdp, W_dp, 2, tid);
    stage_w(sw0,  W_ep, 4, tid);
    stage_w(sw1,  W_eg, 4, tid);
    if (tid >= 32 && tid < 48) {          // nl
        int c = tid - 32;
        float acc = b_nl[c];
        for (int s = 0; s < CS; ++s) acc += nodef[(size_t)n * CS + s] * W_nl[s * CG + c];
        nlr[c] = acc;
    } else if (tid >= 48 && tid < 64) {   // nr
        int c = tid - 48;
        float acc = b_nr[c];
        for (int s = 0; s < CS; ++s) acc += nodef[(size_t)n * CS + s] * W_nr[s * CG + c];
        nlr[16 + c] = acc;
    }
#if defined(HAVE_ASYNC_LDS)
    wait_asynccnt0();
#else
    for (int idx = tid; idx < KKE * CZ; idx += 256)
        edL[idx] = edgef[(size_t)n * KKE * CZ + idx];
#endif
    __syncthreads();

    // -------- LN stats of edge feature rows (from LDS copy) --------
    if (tid < 32) {
        float s = 0.f, s2 = 0.f;
        const float* row = edL + tid * CZ;
        for (int z = 0; z < CZ; ++z) { float x = row[z]; s += x; s2 += x * x; }
        float m = s * (1.0f / CZ);
        float v = s2 * (1.0f / CZ) - m * m;
        stat[tid * 2 + 0] = m;
        stat[tid * 2 + 1] = rsqrtf(v + 1e-5f);
    }
    __syncthreads();

    // -------- normalize ef into LDS ; gate3 --------
    for (int idx = tid; idx < KKE * CZ; idx += 256) {
        int r = idx >> 7, z = idx & 127;
        float x = edL[r * CZ + z];
        efL[r * 132 + z] = (x - stat[r * 2]) * stat[r * 2 + 1] * ln_g[z] + ln_b[z];
    }
    if (tid < CZ) {
        float acc = b_dg[tid];
        for (int ab = 0; ab < 256; ++ab)
            acc += nlr[ab >> 4] * nlr[16 + (ab & 15)] * W_dg[ab * CZ + tid];
        g3[tid] = sigmoidf_(acc);
    }
    __syncthreads();

    // -------- ef A-tiles (held in registers across the gate phases) --------
    v16h aef[2][4];
#pragma unroll
    for (int m = 0; m < 2; ++m)
#pragma unroll
        for (int ks = 0; ks < 4; ++ks)
            aef[m][ks] = gather_a(efL, 132, m, ks, lane);

    // -------- e2 = sigmoid(ef@W_eg+b) * (ef@W_ep+b), wave w owns z-tile w ----
    // (writes e2T, overwriting the raw edge block which is now dead)
    {
        const int zt = w;
        const int z  = zt * 16 + col;
        v8f cep0 = {}, cep1 = {}, ceg0 = {}, ceg1 = {};
#pragma unroll
        for (int ks = 0; ks < 4; ++ks) {
            v16h bep = *(const v16h*)(sw0 + ((ks * 8 + zt) * 32 + lane) * 16);
            v16h beg = *(const v16h*)(sw1 + ((ks * 8 + zt) * 32 + lane) * 16);
            cep0 = wmma16(aef[0][ks], bep, cep0);
            cep1 = wmma16(aef[1][ks], bep, cep1);
            ceg0 = wmma16(aef[0][ks], beg, ceg0);
            ceg1 = wmma16(aef[1][ks], beg, ceg1);
        }
        float bpe = b_ep[z], bge = b_eg[z];
        float spart = 0.f;
#pragma unroll
        for (int v = 0; v < 8; ++v) {
            int j0 = v + 8 * h;
            float e20 = sigmoidf_(ceg0[v] + bge) * (cep0[v] + bpe);
            float e21 = sigmoidf_(ceg1[v] + bge) * (cep1[v] + bpe);
            e2T[z * 40 + j0]      = e20;   // transposed, padded stride 40
            e2T[z * 40 + 16 + j0] = e21;
            spart += e20 + e21;
        }
        spart += __shfl_xor(spart, 16, 32);
        if (lane < 16) se2[z] = spart;     // sum_j e2[j,z]
    }
    __syncthreads();

    // -------- re-stage: W_og, W_lo over the same 64KB region --------
    stage_w(sw0, W_og, 4, tid);
    stage_w(sw1, W_lo, 4, tid);
    __syncthreads();

    // -------- out-gate sigmoid(ef@W_og+b) --------
    {
        const int zt = w;
        const int z  = zt * 16 + col;
        v8f cog0 = {}, cog1 = {};
#pragma unroll
        for (int ks = 0; ks < 4; ++ks) {
            v16h bog = *(const v16h*)(sw0 + ((ks * 8 + zt) * 32 + lane) * 16);
            cog0 = wmma16(aef[0][ks], bog, cog0);
            cog1 = wmma16(aef[1][ks], bog, cog1);
        }
        float bo = b_og[z];
#pragma unroll
        for (int v = 0; v < 8; ++v) {
            int j0 = v + 8 * h;
            ogS[j0 * CZ + z]        = (_Float16)sigmoidf_(cog0[v] + bo);
            ogS[(16 + j0) * CZ + z] = (_Float16)sigmoidf_(cog1[v] + bo);
        }
    }

    // -------- dominant phase: RBF(dist) @ W_dp, e2-weighted reduce over j ----
    // upd[i,z] = g3[z]*( sum_j GEMM[j,z]*e2[j,z] + b_dp[z]*sum_j e2[j,z] )
    const float mustep = 20.0f / 63.0f;
    const float invsig = 64.0f / 20.0f;
    for (int i = w * 4; i < w * 4 + 4; ++i) {
        // Build 4 A-tiles (m=0..1, ks=0..1) of exp(-((d-mu_r)/sigma)^2) in-reg
        v16h ar[2][2];
#pragma unroll
        for (int m = 0; m < 2; ++m) {
            float d = distL[i * 33 + 16 * m + col];
#pragma unroll
            for (int ks = 0; ks < 2; ++ks) {
                v16h t;
#pragma unroll
                for (int e = 0; e < 16; ++e) {
                    int r = kmap(h, e) + 32 * ks;
                    float u = (d - mustep * (float)r) * invsig;
                    t[e] = (_Float16)__expf(-u * u);
                }
                ar[m][ks] = t;
            }
        }
#pragma unroll
        for (int zt = 0; zt < 8; ++zt) {
            v8f c0 = {}, c1 = {};
#pragma unroll
            for (int ks = 0; ks < 2; ++ks) {
                v16h b = *(const v16h*)(swdp + ((ks * 8 + zt) * 32 + lane) * 16);
                c0 = wmma16(ar[0][ks], b, c0);
                c1 = wmma16(ar[1][ks], b, c1);
            }
            int z = zt * 16 + col;
            const float4* p0 = (const float4*)(e2T + z * 40 + 8 * h);
            const float4* p1 = (const float4*)(e2T + z * 40 + 16 + 8 * h);
            float4 ea = p0[0], eb = p0[1], ec = p1[0], ed = p1[1];
            float p = c0[0] * ea.x + c0[1] * ea.y + c0[2] * ea.z + c0[3] * ea.w
                    + c0[4] * eb.x + c0[5] * eb.y + c0[6] * eb.z + c0[7] * eb.w
                    + c1[0] * ec.x + c1[1] * ec.y + c1[2] * ec.z + c1[3] * ec.w
                    + c1[4] * ed.x + c1[5] * ed.y + c1[6] * ed.z + c1[7] * ed.w;
            p += __shfl_xor(p, 16, 32);
            if (lane < 16)
                efL[i * 132 + z] = g3[z] * (p + b_dp[z] * se2[z]); // upd accum
        }
    }
    __syncthreads();

    // -------- LN(upd) in place --------
    if (tid < 32) {
        float s = 0.f, s2 = 0.f;
        const float* row = efL + tid * 132;
        for (int z = 0; z < CZ; ++z) { float x = row[z]; s += x; s2 += x * x; }
        float m = s * (1.0f / CZ);
        float v = s2 * (1.0f / CZ) - m * m;
        stat[tid * 2 + 0] = m;
        stat[tid * 2 + 1] = rsqrtf(v + 1e-5f);
    }
    __syncthreads();
    for (int idx = tid; idx < KKE * CZ; idx += 256) {
        int r = idx >> 7, z = idx & 127;
        efL[r * 132 + z] =
            (efL[r * 132 + z] - stat[r * 2]) * stat[r * 2 + 1] * lno_g[z] + lno_b[z];
    }
    __syncthreads();

    // -------- final: LN(upd) @ W_lo + b_lo, times out-gate, write out --------
    {
        const int zt = w;
        const int z  = zt * 16 + col;
        float bl = b_lo[z];
#pragma unroll
        for (int m = 0; m < 2; ++m) {
            v8f c = {};
#pragma unroll
            for (int ks = 0; ks < 4; ++ks) {
                v16h a = gather_a(efL, 132, m, ks, lane);
                v16h b = *(const v16h*)(sw1 + ((ks * 8 + zt) * 32 + lane) * 16);
                c = wmma16(a, b, c);
            }
#pragma unroll
            for (int v = 0; v < 8; ++v) {
                int row = v + 8 * h + 16 * m;
                float o = (c[v] + bl) * (float)ogS[row * CZ + z];
                out[(size_t)(n * KKE + row) * CZ + z] = o;
            }
        }
    }
}

extern "C" void kernel_launch(void* const* d_in, const int* in_sizes, int n_in,
                              void* d_out, int out_size, void* d_ws, size_t ws_size,
                              hipStream_t stream) {
    const float* nodef  = (const float*)d_in[0];
    const float* ntrans = (const float*)d_in[1];
    const float* edgef  = (const float*)d_in[2];
    const int*   eidx   = (const int*)d_in[3];
    const float* W_nl   = (const float*)d_in[4];
    const float* b_nl   = (const float*)d_in[5];
    const float* W_nr   = (const float*)d_in[6];
    const float* b_nr   = (const float*)d_in[7];
    const float* ln_g   = (const float*)d_in[8];
    const float* ln_b   = (const float*)d_in[9];
    const float* W_ep   = (const float*)d_in[10];
    const float* b_ep   = (const float*)d_in[11];
    const float* W_eg   = (const float*)d_in[12];
    const float* b_eg   = (const float*)d_in[13];
    const float* W_dg   = (const float*)d_in[14];
    const float* b_dg   = (const float*)d_in[15];
    const float* W_dp   = (const float*)d_in[16];
    const float* b_dp   = (const float*)d_in[17];
    const float* lno_g  = (const float*)d_in[18];
    const float* lno_b  = (const float*)d_in[19];
    const float* W_lo   = (const float*)d_in[20];
    const float* b_lo   = (const float*)d_in[21];
    const float* W_og   = (const float*)d_in[22];
    const float* b_og   = (const float*)d_in[23];
    float* out = (float*)d_out;

    (void)hipFuncSetAttribute((const void*)ntmu_kernel,
                              hipFuncAttributeMaxDynamicSharedMemorySize,
                              SHMEM_BYTES);
    ntmu_kernel<<<NN, 256, SHMEM_BYTES, stream>>>(
        nodef, ntrans, edgef, eidx, W_nl, b_nl, W_nr, b_nr, ln_g, ln_b,
        W_ep, b_ep, W_eg, b_eg, W_dg, b_dg, W_dp, b_dp,
        lno_g, lno_b, W_lo, b_lo, W_og, b_og, out);
}
